// TSATransformerBlock_15960098472462
// MI455X (gfx1250) — matrix operand order repaired and checked by reading, not compiled, not run on hardware
//
#include <hip/hip_runtime.h>

#define DEV __device__ __forceinline__

#ifndef __has_builtin
#define __has_builtin(x) 0
#endif
#if __has_builtin(__builtin_amdgcn_global_load_async_to_lds_b128) && \
    __has_builtin(__builtin_amdgcn_s_wait_asynccnt)
#define HAVE_ASYNC_LDS 1
#else
#define HAVE_ASYNC_LDS 0
#endif

#if HAVE_ASYNC_LDS
typedef int v4i_async __attribute__((vector_size(16)));
#define GLOBAL_V4(p) ((__attribute__((address_space(1))) v4i_async*)(void*)(p))
#define LDS_V4(p)    ((__attribute__((address_space(3))) v4i_async*)(void*)(p))
#endif

typedef __attribute__((ext_vector_type(16))) __bf16 v16bf;
typedef __attribute__((ext_vector_type(8)))  float  v8f;

union BfVec { uint4 u[2]; v16bf v; };

DEV unsigned short f2bf(float f) {
  unsigned int u = __float_as_uint(f);
  u += 0x7FFFu + ((u >> 16) & 1u);          // round-to-nearest-even
  return (unsigned short)(u >> 16);
}

DEV v8f vzero8() { v8f z = {0.f,0.f,0.f,0.f,0.f,0.f,0.f,0.f}; return z; }

DEV v8f wmma_bf16(const BfVec& a, const BfVec& b, v8f c) {
  // (neg_a, A, neg_b, B, c_mod, C, reuse_a, reuse_b) -> v_wmma_f32_16x16x32_bf16
  return __builtin_amdgcn_wmma_f32_16x16x32_bf16(false, a.v, false, b.v,
                                                 (short)0, c, false, false);
}

// ---------------------------------------------------------------------------
// fp32 -> bf16 conversion (weights, pos_emb)
// ---------------------------------------------------------------------------
__global__ void k_f32_to_bf16(const float* __restrict__ in,
                              unsigned short* __restrict__ out, int n) {
  int i = blockIdx.x * blockDim.x + threadIdx.x;
  int stride = gridDim.x * blockDim.x;
  for (; i < n; i += stride) out[i] = f2bf(in[i]);
}

// ---------------------------------------------------------------------------
// fused sigmoid-gate + RMSNorm -> bf16   (one block per row, D = 1024)
// ---------------------------------------------------------------------------
__global__ void k_gate_rmsnorm(const float* __restrict__ x,
                               const float* __restrict__ xt,
                               const float* __restrict__ lcw,
                               const float* __restrict__ w,
                               unsigned short* __restrict__ out) {
  __shared__ float red[256];
  const int row = blockIdx.x, tid = threadIdx.x;
  const float lc = 1.0f / (1.0f + __expf(-lcw[0]));
  const float* xr = x  + (size_t)row * 1024;
  const float* tr = xt + (size_t)row * 1024;
  float xa[4]; float ss = 0.f;
#pragma unroll
  for (int j = 0; j < 4; ++j) {
    int c = tid + j * 256;
    float v = lc * xr[c] + (1.0f - lc) * tr[c];
    xa[j] = v; ss += v * v;
  }
  red[tid] = ss; __syncthreads();
  for (int s = 128; s > 0; s >>= 1) { if (tid < s) red[tid] += red[tid + s]; __syncthreads(); }
  const float rms = rsqrtf(red[0] * (1.0f / 1024.0f) + 1e-5f);
  unsigned short* orow = out + (size_t)row * 1024;
#pragma unroll
  for (int j = 0; j < 4; ++j) {
    int c = tid + j * 256;
    orow[c] = f2bf(xa[j] * rms * w[c]);
  }
}

// RMSNorm -> bf16 (no gate)
__global__ void k_rmsnorm(const float* __restrict__ x,
                          const float* __restrict__ w,
                          unsigned short* __restrict__ out) {
  __shared__ float red[256];
  const int row = blockIdx.x, tid = threadIdx.x;
  const float* xr = x + (size_t)row * 1024;
  float xa[4]; float ss = 0.f;
#pragma unroll
  for (int j = 0; j < 4; ++j) {
    int c = tid + j * 256;
    float v = xr[c];
    xa[j] = v; ss += v * v;
  }
  red[tid] = ss; __syncthreads();
  for (int s = 128; s > 0; s >>= 1) { if (tid < s) red[tid] += red[tid + s]; __syncthreads(); }
  const float rms = rsqrtf(red[0] * (1.0f / 1024.0f) + 1e-5f);
  unsigned short* orow = out + (size_t)row * 1024;
#pragma unroll
  for (int j = 0; j < 4; ++j) {
    int c = tid + j * 256;
    orow[c] = f2bf(xa[j] * rms * w[c]);
  }
}

// h = silu(g1) * g3 -> bf16
__global__ void k_silu_mul(const float* __restrict__ g1,
                           const float* __restrict__ g3,
                           unsigned short* __restrict__ out, int n) {
  int i = blockIdx.x * blockDim.x + threadIdx.x;
  int stride = gridDim.x * blockDim.x;
  for (; i < n; i += stride) {
    float a = g1[i];
    float s = a / (1.0f + __expf(-a));
    out[i] = f2bf(s * g3[i]);
  }
}

// ---------------------------------------------------------------------------
// C[M,N] = A[M,K](bf16) @ W[N,K](bf16)^T  (+bias)(+resid) -> fp32 and/or bf16
// block = 256 thr = 8 waves; block tile 128(M) x 128(N); K-step 32
// wave grid 4(M) x 2(N); each wave: 32x64 via 8 accumulators
// staging: double-buffered GLOBAL_LOAD_ASYNC_TO_LDS_B128 — next K-tile
// streams into LDS (ASYNCcnt) while current tile feeds the WMMAs
// ---------------------------------------------------------------------------
__global__ __launch_bounds__(256, 2)
void k_gemm_bf16(const unsigned short* __restrict__ A,
                 const unsigned short* __restrict__ W,
                 const float* __restrict__ bias,
                 const float* __restrict__ resid,
                 float* __restrict__ outF,
                 unsigned short* __restrict__ outB,
                 int M, int N, int K) {
  __shared__ unsigned short As[2][128][32];
  __shared__ unsigned short Bs[2][128][32];
  const int tid  = threadIdx.x;
  const int lane = tid & 31, wid = tid >> 5;
  const int g = lane >> 4, ln = lane & 15;
  const int rowBase = blockIdx.y * 128;
  const int colBase = blockIdx.x * 128;
  const int wm = wid >> 1, wn = wid & 1;          // 4 x 2 wave grid
  const int m0 = wm * 32, n0 = wn * 64;

  v8f c[2][4];
#pragma unroll
  for (int mi = 0; mi < 2; ++mi)
#pragma unroll
    for (int ni = 0; ni < 4; ++ni) c[mi][ni] = vzero8();

  const int ar = tid >> 1;            // 0..127  (same mapping for As and Bs)
  const int ac = (tid & 1) * 16;      // 0 or 16

  const unsigned short* gA = A + (size_t)(rowBase + ar) * K + ac;
  const unsigned short* gW = W + (size_t)(colBase + ar) * K + ac;

  // stage K-tile `k0` into LDS buffer `bb`
  auto stage = [&](int bb, int k0) {
#if HAVE_ASYNC_LDS
    __builtin_amdgcn_global_load_async_to_lds_b128(GLOBAL_V4(gA + k0),
                                                   LDS_V4(&As[bb][ar][ac]), 0, 0);
    __builtin_amdgcn_global_load_async_to_lds_b128(GLOBAL_V4(gA + k0 + 8),
                                                   LDS_V4(&As[bb][ar][ac + 8]), 0, 0);
    __builtin_amdgcn_global_load_async_to_lds_b128(GLOBAL_V4(gW + k0),
                                                   LDS_V4(&Bs[bb][ar][ac]), 0, 0);
    __builtin_amdgcn_global_load_async_to_lds_b128(GLOBAL_V4(gW + k0 + 8),
                                                   LDS_V4(&Bs[bb][ar][ac + 8]), 0, 0);
#else
    const uint4* pa = (const uint4*)(gA + k0);
    const uint4* pw = (const uint4*)(gW + k0);
    uint4 t0 = pa[0], t1 = pa[1], t2 = pw[0], t3 = pw[1];
    *(uint4*)&As[bb][ar][ac]     = t0;
    *(uint4*)&As[bb][ar][ac + 8] = t1;
    *(uint4*)&Bs[bb][ar][ac]     = t2;
    *(uint4*)&Bs[bb][ar][ac + 8] = t3;
#endif
  };

  stage(0, 0);                        // prologue: tile 0 in flight
  int buf = 0;

  for (int k0 = 0; k0 < K; k0 += 32, buf ^= 1) {
#if HAVE_ASYNC_LDS
    __builtin_amdgcn_s_wait_asynccnt(0);   // tile k0 resident in LDS[buf]
#endif
    __syncthreads();                        // prior tile's readers drained (DScnt)
    if (k0 + 32 < K) {
      stage(buf ^ 1, k0 + 32);              // stream next tile during compute
      if (k0 + 64 < K) {                    // global_prefetch_b8 two tiles ahead
        __builtin_prefetch(gA + k0 + 64, 0, 1);
        __builtin_prefetch(gW + k0 + 64, 0, 1);
      }
    }

    BfVec a0, a1;
    // A layout (16x32 bf16): lane m=ln, group g; k = 16*(j>>3) + 8g + (j&7)
    a0.u[0] = *(const uint4*)&As[buf][m0 + ln][8 * g];
    a0.u[1] = *(const uint4*)&As[buf][m0 + ln][16 + 8 * g];
    a1.u[0] = *(const uint4*)&As[buf][m0 + 16 + ln][8 * g];
    a1.u[1] = *(const uint4*)&As[buf][m0 + 16 + ln][16 + 8 * g];
#pragma unroll
    for (int ni = 0; ni < 4; ++ni) {
      BfVec b;
      // B layout (32x16 bf16): lane n=ln, group g; k = 16*(j>>3) + 8g + (j&7)
      b.u[0] = *(const uint4*)&Bs[buf][n0 + ni * 16 + ln][8 * g];
      b.u[1] = *(const uint4*)&Bs[buf][n0 + ni * 16 + ln][16 + 8 * g];
      c[0][ni] = wmma_bf16(a0, b, c[0][ni]);
      c[1][ni] = wmma_bf16(a1, b, c[1][ni]);
    }
  }

#pragma unroll
  for (int mi = 0; mi < 2; ++mi)
#pragma unroll
    for (int ni = 0; ni < 4; ++ni) {
      const int tr0 = rowBase + m0 + mi * 16 + 8 * g;     // C/D: vgpr r -> M = 8g+r
      const int tc  = colBase + n0 + ni * 16 + ln;        //       lane  -> N
#pragma unroll
      for (int r = 0; r < 8; ++r) {
        const size_t idx = (size_t)(tr0 + r) * N + tc;
        float v = c[mi][ni][r];
        if (bias)  v += bias[tc];
        if (resid) v += resid[idx];
        if (outF)  outF[idx] = v;
        if (outB)  outB[idx] = f2bf(v);
      }
    }
}

// ---------------------------------------------------------------------------
// flash attention: grid (T/128, H, B), block 256 = 8 waves
// each wave owns 16 query rows; K/V streamed in chunks of 32 keys via LDS
// ---------------------------------------------------------------------------
__global__ __launch_bounds__(256, 2)
void k_attention(const unsigned short* __restrict__ Q,
                 const unsigned short* __restrict__ Kb,
                 const unsigned short* __restrict__ V,
                 unsigned short* __restrict__ O) {
  __shared__ unsigned short Ks[32][64];       // [key][hd]
  __shared__ unsigned short Vt[64][32];       // [hd][key] (transposed)
  __shared__ unsigned short Ps[8][16][32];    // per-wave P tile [qrow][key]
  const int tid  = threadIdx.x;
  const int lane = tid & 31, w = tid >> 5;
  const int g = lane >> 4, ln = lane & 15;
  const int h = blockIdx.y, b = blockIdx.z;
  const int qbase = blockIdx.x * 128;
  const int T = 2048, D = 1024;
  const size_t bT = (size_t)b * T;
  const int qrow0 = qbase + w * 16;

  // Q tile in A layout, loaded once (two HD-halves of 32)
  const unsigned short* qr = Q + (bT + qrow0 + ln) * (size_t)D + h * 64;
  BfVec aq0, aq1;
  aq0.u[0] = *(const uint4*)(qr + 16 * g);
  aq0.u[1] = *(const uint4*)(qr + 16 * g + 8);
  aq1.u[0] = *(const uint4*)(qr + 32 + 16 * g);
  aq1.u[1] = *(const uint4*)(qr + 32 + 16 * g + 8);

  v8f acc[4];
  float runmax[8], runsum[8];
#pragma unroll
  for (int t = 0; t < 4; ++t) acc[t] = vzero8();
#pragma unroll
  for (int r = 0; r < 8; ++r) { runmax[r] = -3.0e38f; runsum[r] = 0.f; }

  const int nch  = (qbase + 128) >> 5;      // uniform over block (causal)
  const int krow = tid >> 3;                // 0..31
  const int kc8  = (tid & 7) * 8;           // hd chunk of 8

  for (int ch = 0; ch < nch; ++ch) {
    const int kcb = ch * 32;
    // cooperative K chunk load (async direct-to-LDS when available)
#if HAVE_ASYNC_LDS
    __builtin_amdgcn_global_load_async_to_lds_b128(
        GLOBAL_V4(Kb + (size_t)(kcb + krow) * D + h * 64 + kc8),
        LDS_V4(&Ks[krow][kc8]), 0, 0);
#else
    *(uint4*)&Ks[krow][kc8] =
        *(const uint4*)(Kb + (size_t)(kcb + krow) * D + h * 64 + kc8);
#endif
    // V chunk load with on-the-fly transpose (needs VGPRs, stays synchronous)
    union { uint4 q; unsigned short s[8]; } vv;
    vv.q = *(const uint4*)(V + (bT + kcb + krow) * (size_t)D + h * 64 + kc8);
#pragma unroll
    for (int e = 0; e < 8; ++e) Vt[kc8 + e][krow] = vv.s[e];
#if HAVE_ASYNC_LDS
    __builtin_amdgcn_s_wait_asynccnt(0);
#endif
    __syncthreads();

    // scores (16q x 32k): two 16x16 tiles, each = chained WMMA over HD=64
    v8f s0 = vzero8(), s1 = vzero8();
    BfVec bk;
    bk.u[0] = *(const uint4*)&Ks[ln][16 * g];
    bk.u[1] = *(const uint4*)&Ks[ln][16 * g + 8];
    s0 = wmma_bf16(aq0, bk, s0);
    bk.u[0] = *(const uint4*)&Ks[ln][32 + 16 * g];
    bk.u[1] = *(const uint4*)&Ks[ln][32 + 16 * g + 8];
    s0 = wmma_bf16(aq1, bk, s0);
    bk.u[0] = *(const uint4*)&Ks[16 + ln][16 * g];
    bk.u[1] = *(const uint4*)&Ks[16 + ln][16 * g + 8];
    s1 = wmma_bf16(aq0, bk, s1);
    bk.u[0] = *(const uint4*)&Ks[16 + ln][32 + 16 * g];
    bk.u[1] = *(const uint4*)&Ks[16 + ln][32 + 16 * g + 8];
    s1 = wmma_bf16(aq1, bk, s1);

    // scale (D^-0.5 = 1/32), causal mask, online softmax update
    const int col0 = kcb + ln, col1 = kcb + 16 + ln;
#pragma unroll
    for (int r = 0; r < 8; ++r) {
      const int qrow = qrow0 + 8 * g + r;
      float v0 = s0[r] * 0.03125f;
      float v1 = s1[r] * 0.03125f;
      if (col0 > qrow) v0 = -3.0e38f;
      if (col1 > qrow) v1 = -3.0e38f;
      float cm = fmaxf(v0, v1);
      cm = fmaxf(cm, __shfl_xor(cm, 1));
      cm = fmaxf(cm, __shfl_xor(cm, 2));
      cm = fmaxf(cm, __shfl_xor(cm, 4));
      cm = fmaxf(cm, __shfl_xor(cm, 8));
      const float nm   = fmaxf(runmax[r], cm);
      const float corr = __expf(runmax[r] - nm);
      const float p0 = __expf(v0 - nm);
      const float p1 = __expf(v1 - nm);
      float rs = p0 + p1;
      rs += __shfl_xor(rs, 1);
      rs += __shfl_xor(rs, 2);
      rs += __shfl_xor(rs, 4);
      rs += __shfl_xor(rs, 8);
      runsum[r] = runsum[r] * corr + rs;
      runmax[r] = nm;
#pragma unroll
      for (int t = 0; t < 4; ++t) acc[t][r] *= corr;
      Ps[w][8 * g + r][ln]      = f2bf(p0);
      Ps[w][8 * g + r][16 + ln] = f2bf(p1);
    }
    asm volatile("" ::: "memory");   // keep P store->load order (same-wave LDS is in-order)

    // P (A layout) x V (B layout) -> acc over 4 HD column tiles
    BfVec pa, bv;
    pa.u[0] = *(const uint4*)&Ps[w][ln][16 * g];
    pa.u[1] = *(const uint4*)&Ps[w][ln][16 * g + 8];
#pragma unroll
    for (int t = 0; t < 4; ++t) {
      bv.u[0] = *(const uint4*)&Vt[t * 16 + ln][16 * g];
      bv.u[1] = *(const uint4*)&Vt[t * 16 + ln][16 * g + 8];
      acc[t] = wmma_bf16(pa, bv, acc[t]);
    }
    __syncthreads();
  }

#pragma unroll
  for (int t = 0; t < 4; ++t)
#pragma unroll
    for (int r = 0; r < 8; ++r) {
      const int qrow = qrow0 + 8 * g + r;
      O[(bT + qrow) * (size_t)D + h * 64 + t * 16 + ln] = f2bf(acc[t][r] / runsum[r]);
    }
}

// ---------------------------------------------------------------------------
extern "C" void kernel_launch(void* const* d_in, const int* in_sizes, int n_in,
                              void* d_out, int out_size, void* d_ws, size_t ws_size,
                              hipStream_t stream) {
  (void)in_sizes; (void)n_in; (void)out_size; (void)ws_size;
  const float* x     = (const float*)d_in[0];
  const float* x_tok = (const float*)d_in[1];
  const float* pos   = (const float*)d_in[2];
  const float* lcw   = (const float*)d_in[3];
  const float* Wq    = (const float*)d_in[4];
  const float* Wk    = (const float*)d_in[5];
  const float* Wv    = (const float*)d_in[6];
  const float* projW = (const float*)d_in[7];
  const float* projB = (const float*)d_in[8];
  const float* w1W   = (const float*)d_in[9];
  const float* w1Bv  = (const float*)d_in[10];
  const float* w2W   = (const float*)d_in[11];
  const float* w2Bv  = (const float*)d_in[12];
  const float* w3W   = (const float*)d_in[13];
  const float* w3Bv  = (const float*)d_in[14];
  const float* n1w   = (const float*)d_in[15];
  const float* n2w   = (const float*)d_in[16];

  constexpr int Bx = 2, T = 2048, D = 1024, H = 16, FF = 4096, M = Bx * T;

  char* base = (char*)d_ws;
  size_t off = 0;
  auto alloc = [&](size_t bytes) -> char* {
    char* p = base + off;
    off += (bytes + 255) & ~(size_t)255;
    return p;
  };
  unsigned short* wqB  = (unsigned short*)alloc((size_t)D * D * 2);
  unsigned short* wkB  = (unsigned short*)alloc((size_t)D * D * 2);
  unsigned short* wvB  = (unsigned short*)alloc((size_t)D * D * 2);
  unsigned short* wpB  = (unsigned short*)alloc((size_t)D * D * 2);
  unsigned short* w1Bf = (unsigned short*)alloc((size_t)FF * D * 2);
  unsigned short* w2Bf = (unsigned short*)alloc((size_t)D * FF * 2);
  unsigned short* w3Bf = (unsigned short*)alloc((size_t)FF * D * 2);
  unsigned short* posB = (unsigned short*)alloc((size_t)T * D * 2);
  unsigned short* xnB  = (unsigned short*)alloc((size_t)M * D * 2);
  unsigned short* qB   = (unsigned short*)alloc((size_t)M * D * 2);
  unsigned short* kB   = (unsigned short*)alloc((size_t)T * D * 2);
  unsigned short* vB   = (unsigned short*)alloc((size_t)M * D * 2);
  unsigned short* atB  = (unsigned short*)alloc((size_t)M * D * 2);
  float*          x1   = (float*)alloc((size_t)M * D * 4);
  unsigned short* xn2B = (unsigned short*)alloc((size_t)M * D * 2);
  float*          g1   = (float*)alloc((size_t)M * FF * 4);
  float*          g3   = (float*)alloc((size_t)M * FF * 4);
  unsigned short* hB   = (unsigned short*)alloc((size_t)M * FF * 2);

  const dim3 blk(256);
  auto cvt = [&](const float* s, unsigned short* d, int n) {
    k_f32_to_bf16<<<dim3((n + 1023) / 1024), blk, 0, stream>>>(s, d, n);
  };
  cvt(Wq, wqB, D * D);
  cvt(Wk, wkB, D * D);
  cvt(Wv, wvB, D * D);
  cvt(projW, wpB, D * D);
  cvt(w1W, w1Bf, FF * D);
  cvt(w2W, w2Bf, D * FF);
  cvt(w3W, w3Bf, FF * D);
  cvt(pos, posB, T * D);

  k_gate_rmsnorm<<<dim3(M), blk, 0, stream>>>(x, x_tok, lcw, n1w, xnB);

  k_gemm_bf16<<<dim3(D / 128, M / 128), blk, 0, stream>>>(xnB,  wqB, nullptr, nullptr, nullptr, qB, M, D, D);
  k_gemm_bf16<<<dim3(D / 128, T / 128), blk, 0, stream>>>(posB, wkB, nullptr, nullptr, nullptr, kB, T, D, D);
  k_gemm_bf16<<<dim3(D / 128, M / 128), blk, 0, stream>>>(xnB,  wvB, nullptr, nullptr, nullptr, vB, M, D, D);

  k_attention<<<dim3(T / 128, H, Bx), blk, 0, stream>>>(qB, kB, vB, atB);

  k_gemm_bf16<<<dim3(D / 128, M / 128), blk, 0, stream>>>(atB, wpB, projB, x, x1, nullptr, M, D, D);

  k_rmsnorm<<<dim3(M), blk, 0, stream>>>(x1, n2w, xn2B);

  k_gemm_bf16<<<dim3(FF / 128, M / 128), blk, 0, stream>>>(xn2B, w1Bf, w1Bv, nullptr, g1, nullptr, M, FF, D);
  k_gemm_bf16<<<dim3(FF / 128, M / 128), blk, 0, stream>>>(xn2B, w3Bf, w3Bv, nullptr, g3, nullptr, M, FF, D);
  k_silu_mul<<<dim3((M * FF + 1023) / 1024), blk, 0, stream>>>(g1, g3, hB, M * FF);

  k_gemm_bf16<<<dim3(D / 128, M / 128), blk, 0, stream>>>(hB, w2Bf, w2Bv, x1, (float*)d_out, nullptr, M, D, FF);
}
